// BaselineGAT_28346784153662
// MI455X (gfx1250) — matrix operand order repaired and checked by reading, not compiled, not run on hardware
//
#include <hip/hip_runtime.h>
#include <hip/hip_bf16.h>

typedef __attribute__((ext_vector_type(16))) _Float16 v16h;
typedef __attribute__((ext_vector_type(8)))  _Float16 v8h;
typedef __attribute__((ext_vector_type(8)))  float    v8f;

#define NEG_SLOPE 0.2f

// ---------------------------------------------------------------------------
// Convert f32 -> f16, 4 elements per thread (vectorized read).
// ---------------------------------------------------------------------------
__global__ void gat_cvt_f16(const float* __restrict__ in, _Float16* __restrict__ out,
                            long long n4)
{
    long long i = (long long)blockIdx.x * blockDim.x + threadIdx.x;
    if (i >= n4) return;
    const float4 v = *(const float4*)(in + 4 * i);
    _Float16* o = out + 4 * i;
    o[0] = (_Float16)v.x; o[1] = (_Float16)v.y;
    o[2] = (_Float16)v.z; o[3] = (_Float16)v.w;
}

// ---------------------------------------------------------------------------
// Pack W[K,N] (f32 row-major) into WMMA B-fragment-ready f16 layout:
// dword index = ((kt*(N/16)+nt)*32 + lane)*8 + v  holds B rows K=2(v&3)+16(v>>2),+1
// (+8 for lanes 16..31) of column nt*16+(lane&15), per ISA 7.12.2.
// ---------------------------------------------------------------------------
__global__ void gat_pack_bfrag(const float* __restrict__ W, unsigned* __restrict__ P,
                               int K, int N)
{
    int i = blockIdx.x * blockDim.x + threadIdx.x;
    int total = (K / 32) * (N / 16) * 256;
    if (i >= total) return;
    int v    = i & 7;
    int lane = (i >> 3) & 31;
    int frag = i >> 8;
    int ntN  = N >> 4;
    int kt = frag / ntN, nt = frag % ntN;
    int kb  = kt * 32 + ((lane >> 4) * 8) + 2 * (v & 3) + ((v >> 2) * 16);
    int col = nt * 16 + (lane & 15);
    unsigned short lo = __builtin_bit_cast(unsigned short, (_Float16)W[(size_t)kb * N + col]);
    unsigned short hi = __builtin_bit_cast(unsigned short, (_Float16)W[(size_t)(kb + 1) * N + col]);
    P[i] = (unsigned)lo | ((unsigned)hi << 16);
}

// ---------------------------------------------------------------------------
// WMMA GEMM, K fixed at 128 (both layers): C[M,N] = A16[M,128] @ B (packed).
// One wave per 16x16 C tile; 4 fully-unrolled v_wmma_f32_16x16x32_f16.
// A fragment: two contiguous b128 loads per lane. B fragment: one 32B load.
// ---------------------------------------------------------------------------
__global__ __launch_bounds__(256) void gat_gemm_wmma_k128(
    const _Float16* __restrict__ A16, const unsigned* __restrict__ BP,
    float* __restrict__ C, int M, int N)
{
    constexpr int K = 128;
    const int lane = threadIdx.x & 31;
    const int wave = threadIdx.x >> 5;
    const int ntN  = N >> 4;
    const int tile = blockIdx.x * 8 + wave;
    if (tile >= (M >> 4) * ntN) return;          // wave-uniform guard
    const int mt = tile / ntN;
    const int nt = tile % ntN;

    const int arow  = mt * 16 + (lane & 15);
    const int bcol  = nt * 16 + (lane & 15);
    const int koffA = (lane >> 4) * 8;           // lanes 16-31: K base +8

    const _Float16* ap = A16 + (size_t)arow * K + koffA;
    v8f acc = {};
    #pragma unroll
    for (int kt = 0; kt < K / 32; ++kt) {
        const v8h alo = *(const v8h*)(ap + kt * 32);        // K = kt*32 + koffA + 0..7
        const v8h ahi = *(const v8h*)(ap + kt * 32 + 16);   // K = ... + 16..23
        const v16h a = __builtin_shufflevector(alo, ahi,
            0, 1, 2, 3, 4, 5, 6, 7, 8, 9, 10, 11, 12, 13, 14, 15);
        const v16h b = *(const v16h*)(BP + ((size_t)(kt * ntN + nt) * 32 + lane) * 8);
        acc = __builtin_amdgcn_wmma_f32_16x16x32_f16(
                  false, a, false, b, (short)0, acc, false, false);
    }
    // C/D layout: VGPR r -> row r (lanes 0-15) / row r+8 (lanes 16-31)
    const int rbase = mt * 16 + ((lane >> 4) << 3);
    #pragma unroll
    for (int r = 0; r < 8; ++r)
        C[(size_t)(rbase + r) * N + bcol] = acc[r];
}

// ---------------------------------------------------------------------------
// Per-(node,head) attention coefficients: a_src/a_dst = <h[n,h,:], att[h,:]>
// ---------------------------------------------------------------------------
template<int HEADS, int DIM>
__global__ void gat_att_coef(const float* __restrict__ h,
                             const float* __restrict__ att_s,
                             const float* __restrict__ att_d,
                             float* __restrict__ a_s, float* __restrict__ a_d,
                             int N)
{
    int i = blockIdx.x * blockDim.x + threadIdx.x;   // i = n*HEADS + hh
    if (i >= N * HEADS) return;
    int hh = i % HEADS;
    const float* hp = h + (size_t)i * DIM;
    const float* sp = att_s + hh * DIM;
    const float* dp = att_d + hh * DIM;
    float s = 0.f, d = 0.f;
    #pragma unroll
    for (int c = 0; c < DIM; ++c) { float v = hp[c]; s += v * sp[c]; d += v * dp[c]; }
    a_s[i] = s; a_d[i] = d;
}

// ---------------------------------------------------------------------------
// Edge pass 1: denom[dst,h] += exp(leaky(a_src[src,h] + a_dst[dst,h]))
// Edges [0,E) from edge_index, [E, E+N) are self loops.
// ---------------------------------------------------------------------------
template<int HEADS>
__global__ void gat_edge_denom(const int* __restrict__ src,
                               const int* __restrict__ dst,
                               const float* __restrict__ a_s,
                               const float* __restrict__ a_d,
                               float* __restrict__ denom,
                               int E, int N)
{
    long long i = (long long)blockIdx.x * blockDim.x + threadIdx.x;
    if (i >= (long long)(E + N) * HEADS) return;
    int e  = (int)(i / HEADS);
    int hh = (int)(i % HEADS);
    int s = (e < E) ? src[e] : (e - E);
    int d = (e < E) ? dst[e] : (e - E);
    float x = a_s[s * HEADS + hh] + a_d[d * HEADS + hh];
    x = (x > 0.f) ? x : NEG_SLOPE * x;
    atomicAdd(&denom[d * HEADS + hh], __expf(x));
}

// ---------------------------------------------------------------------------
// Edge pass 2: alpha[e,h] = exp(leaky(...)) / denom[dst,h]
// ---------------------------------------------------------------------------
template<int HEADS>
__global__ void gat_edge_alpha(const int* __restrict__ src,
                               const int* __restrict__ dst,
                               const float* __restrict__ a_s,
                               const float* __restrict__ a_d,
                               const float* __restrict__ denom,
                               float* __restrict__ alpha,
                               int E, int N)
{
    long long i = (long long)blockIdx.x * blockDim.x + threadIdx.x;
    if (i >= (long long)(E + N) * HEADS) return;
    int e  = (int)(i / HEADS);
    int hh = (int)(i % HEADS);
    int s = (e < E) ? src[e] : (e - E);
    int d = (e < E) ? dst[e] : (e - E);
    float x = a_s[s * HEADS + hh] + a_d[d * HEADS + hh];
    x = (x > 0.f) ? x : NEG_SLOPE * x;
    alpha[i] = __expf(x) / denom[d * HEADS + hh];
}

// ---------------------------------------------------------------------------
// Edge pass 3: out[dst, c..c+3] += alpha[e, head] * h[src, c..c+3]
// One thread per 4 channels: float4 gather + 4 f32 atomic adds.
// ---------------------------------------------------------------------------
template<int HEADS, int DIM>
__global__ void gat_edge_aggregate(const int* __restrict__ src,
                                   const int* __restrict__ dst,
                                   const float* __restrict__ alpha,
                                   const float* __restrict__ h,
                                   float* __restrict__ out,
                                   int E, int N)
{
    constexpr int F  = HEADS * DIM;
    constexpr int C4 = F / 4;
    long long i = (long long)blockIdx.x * blockDim.x + threadIdx.x;
    if (i >= (long long)(E + N) * C4) return;
    int e  = (int)(i / C4);
    int c  = ((int)(i % C4)) * 4;
    int s = (e < E) ? src[e] : (e - E);
    int d = (e < E) ? dst[e] : (e - E);
    const float al = alpha[(long long)e * HEADS + c / DIM];
    const float4 hv = *(const float4*)(h + (size_t)s * F + c);
    float* op = out + (size_t)d * F + c;
    atomicAdd(op + 0, al * hv.x);
    atomicAdd(op + 1, al * hv.y);
    atomicAdd(op + 2, al * hv.z);
    atomicAdd(op + 3, al * hv.w);
}

// ---------------------------------------------------------------------------
// Layer-1 epilogue: hact16 = f16(elu(agg + b1))  (f16 feeds GEMM2 directly)
// ---------------------------------------------------------------------------
template<int F>
__global__ void gat_bias_elu_f16(const float* __restrict__ x,
                                 const float* __restrict__ bias,
                                 _Float16* __restrict__ o, long long total)
{
    long long i = (long long)blockIdx.x * blockDim.x + threadIdx.x;
    if (i >= total) return;
    float v = x[i] + bias[(int)(i & (F - 1))];
    o[i] = (_Float16)((v > 0.f) ? v : (__expf(v) - 1.f));
}

// ---------------------------------------------------------------------------
// out[n,c] = bias[c]  (layer-2 output init; aggregation atomically adds on top)
// ---------------------------------------------------------------------------
template<int F>
__global__ void gat_out_bias_init(float* __restrict__ out,
                                  const float* __restrict__ bias, long long total)
{
    long long i = (long long)blockIdx.x * blockDim.x + threadIdx.x;
    if (i >= total) return;
    out[i] = bias[(int)(i & (F - 1))];
}

static inline unsigned gridFor(long long total, int block)
{
    return (unsigned)((total + block - 1) / block);
}

extern "C" void kernel_launch(void* const* d_in, const int* in_sizes, int n_in,
                              void* d_out, int out_size, void* d_ws, size_t ws_size,
                              hipStream_t stream)
{
    (void)in_sizes; (void)n_in; (void)out_size; (void)ws_size;
    const float* x   = (const float*)d_in[0];
    const int*   ei  = (const int*)d_in[1];     // int32 (JAX x64 disabled)
    const float* W1  = (const float*)d_in[2];
    const float* as1 = (const float*)d_in[3];
    const float* ad1 = (const float*)d_in[4];
    const float* b1  = (const float*)d_in[5];
    const float* W2  = (const float*)d_in[6];
    const float* as2 = (const float*)d_in[7];
    const float* ad2 = (const float*)d_in[8];
    const float* b2  = (const float*)d_in[9];
    float* out = (float*)d_out;

    const int N = 50000, E = 800000;
    const int F1 = 128, H = 4, D1 = 32, F2 = 32;
    const int EP = E + N;                       // edges incl. self-loops
    const int* src = ei;                        // edge_index[0]
    const int* dst = ei + E;                    // edge_index[1]

    // ---- workspace carve-up (32B alignment preserved throughout) ----
    float* p    = (float*)d_ws;
    float* h1   = p; p += (size_t)N * F1;
    float* aS1  = p; p += (size_t)N * H;
    float* aD1  = p; p += (size_t)N * H;
    float* den1 = p; p += (size_t)N * H;
    float* al1  = p; p += (size_t)EP * H;
    float* agg1 = p; p += (size_t)N * F1;
    float* h2   = p; p += (size_t)N * F2;
    float* aS2  = p; p += (size_t)N;
    float* aD2  = p; p += (size_t)N;
    float* den2 = p; p += (size_t)N;
    float* al2  = p; p += (size_t)EP;
    _Float16* xh     = (_Float16*)p;            // N*F1 halves
    _Float16* hact16 = xh + (size_t)N * F1;     // N*F1 halves
    unsigned* BP1 = (unsigned*)(hact16 + (size_t)N * F1);   // 8192 dwords
    unsigned* BP2 = BP1 + (F1 / 32) * (F1 / 16) * 256;      // 2048 dwords

    const int BLK = 256;

    // zero the accumulation buffers (fresh every call; graph-capture safe)
    hipMemsetAsync(den1, 0, sizeof(float) * (size_t)N * H, stream);
    hipMemsetAsync(agg1, 0, sizeof(float) * (size_t)N * F1, stream);
    hipMemsetAsync(den2, 0, sizeof(float) * (size_t)N, stream);

    // prepasses: f16 copy of x, fragment-packed weights
    gat_cvt_f16<<<gridFor((long long)N * F1 / 4, BLK), BLK, 0, stream>>>(
        x, xh, (long long)N * F1 / 4);
    gat_pack_bfrag<<<gridFor((F1 / 32) * (F1 / 16) * 256, BLK), BLK, 0, stream>>>(
        W1, BP1, F1, F1);
    gat_pack_bfrag<<<gridFor((F1 / 32) * (F2 / 16) * 256, BLK), BLK, 0, stream>>>(
        W2, BP2, F1, F2);

    // ---- Layer 1 ----
    {   // h1 = x @ W1 : [50000,128]x[128,128]
        int tiles = (N / 16) * (F1 / 16);
        gat_gemm_wmma_k128<<<(tiles + 7) / 8, 256, 0, stream>>>(xh, BP1, h1, N, F1);
    }
    gat_att_coef<H, D1><<<gridFor((long long)N * H, BLK), BLK, 0, stream>>>(
        h1, as1, ad1, aS1, aD1, N);
    gat_edge_denom<H><<<gridFor((long long)EP * H, BLK), BLK, 0, stream>>>(
        src, dst, aS1, aD1, den1, E, N);
    gat_edge_alpha<H><<<gridFor((long long)EP * H, BLK), BLK, 0, stream>>>(
        src, dst, aS1, aD1, den1, al1, E, N);
    gat_edge_aggregate<H, D1><<<gridFor((long long)EP * (F1 / 4), BLK), BLK, 0, stream>>>(
        src, dst, al1, h1, agg1, E, N);
    gat_bias_elu_f16<F1><<<gridFor((long long)N * F1, BLK), BLK, 0, stream>>>(
        agg1, b1, hact16, (long long)N * F1);

    // ---- Layer 2 ----
    {   // h2 = elu_act @ W2 : [50000,128]x[128,32]
        int tiles = (N / 16) * (F2 / 16);
        gat_gemm_wmma_k128<<<(tiles + 7) / 8, 256, 0, stream>>>(hact16, BP2, h2, N, F2);
    }
    gat_att_coef<1, F2><<<gridFor((long long)N, BLK), BLK, 0, stream>>>(
        h2, as2, ad2, aS2, aD2, N);
    gat_edge_denom<1><<<gridFor((long long)EP, BLK), BLK, 0, stream>>>(
        src, dst, aS2, aD2, den2, E, N);
    gat_edge_alpha<1><<<gridFor((long long)EP, BLK), BLK, 0, stream>>>(
        src, dst, aS2, aD2, den2, al2, E, N);
    gat_out_bias_init<F2><<<gridFor((long long)N * F2, BLK), BLK, 0, stream>>>(
        out, b2, (long long)N * F2);
    gat_edge_aggregate<1, F2><<<gridFor((long long)EP * (F2 / 4), BLK), BLK, 0, stream>>>(
        src, dst, al2, h2, out, E, N);
}
